// LineAttention_58007828300166
// MI455X (gfx1250) — matrix operand order repaired and checked by reading, not compiled, not run on hardware
//
#include <hip/hip_runtime.h>
#include <hip/hip_bf16.h>

#define DIM    256
#define LINE   24
#define HEADS  8
#define DHEAD  64
#define INNER  512
#define NQKV   1536
#define NB     8192

typedef __attribute__((ext_vector_type(16))) __bf16 bf16x16;
typedef __attribute__((ext_vector_type(8)))  float  f32x8;
typedef __attribute__((ext_vector_type(4))) unsigned int u32x4;
typedef __attribute__((ext_vector_type(8)))  int i32x8;
typedef __attribute__((ext_vector_type(4)))  int i32x4;

// LDS geometry (bf16 element pitches)
#define PX   264                 // sX  [32][256] + pad
#define PQ   1544                // sQKV[32][1536] + pad
#define PV   40                  // sVt [512][32] + pad
#define SX_BYTES   (32 * PX * 2)          // 16896
#define SQKV_BYTES (32 * PQ * 2)          // 98816
#define SVT_BYTES  (512 * PV * 2)         // 40960  (also x fp32 staging: 24576B)
#define LDS_BYTES  (SX_BYTES + SQKV_BYTES + SVT_BYTES)

union FragU { uint4 q[2]; bf16x16 v; };

// A operand: [M][K] row-major.  Lane (m = lane&15, h = lane>>4) holds
// K = [k0+16h, +16) sequential -> two adjacent 16B loads.
__device__ __forceinline__ bf16x16 frag_a(const __bf16* base, int row0, int pitch,
                                          int k0, int lane) {
  const __bf16* p = base + (row0 + (lane & 15)) * pitch + k0 + ((lane >> 4) << 4);
  FragU u;
  u.q[0] = *(const uint4*)(p);
  u.q[1] = *(const uint4*)(p + 8);
  return u.v;
}

// B operand from "Bt" layout [N][K] row-major.  Lane (n = lane&15, h = lane>>4)
// holds K = [k0+16h, +16) sequential -> two adjacent 16B loads.
__device__ __forceinline__ bf16x16 frag_b(const __bf16* base, int row0, int pitch,
                                          int k0, int lane) {
  const __bf16* p = base + (row0 + (lane & 15)) * pitch + k0 + ((lane >> 4) << 4);
  FragU u;
  u.q[0] = *(const uint4*)(p);
  u.q[1] = *(const uint4*)(p + 8);
  return u.v;
}

__device__ __forceinline__ f32x8 wmma_bf16(bf16x16 a, bf16x16 b, f32x8 c) {
  return __builtin_amdgcn_wmma_f32_16x16x32_bf16(false, a, false, b, (short)0, c,
                                                 false, false);
}

__device__ __forceinline__ unsigned pk2(float lo, float hi) {
  unsigned short l = __builtin_bit_cast(unsigned short, (__bf16)lo);
  unsigned short h = __builtin_bit_cast(unsigned short, (__bf16)hi);
  return (unsigned)l | ((unsigned)h << 16);
}

__device__ __forceinline__ uint4 pack8(const f32x8& v, float s) {
  uint4 r;
  r.x = pk2(v[0] * s, v[1] * s);
  r.y = pk2(v[2] * s, v[3] * s);
  r.z = pk2(v[4] * s, v[5] * s);
  r.w = pk2(v[6] * s, v[7] * s);
  return r;
}

__device__ __forceinline__ f32x8 zero8() {
  f32x8 z = {0.f, 0.f, 0.f, 0.f, 0.f, 0.f, 0.f, 0.f};
  return z;
}

// ---------------------------------------------------------------------------
// Prep: transpose weights to bf16 "Bt"/"A" friendly layouts in workspace.
//   WtQKV[n][k], n<1536, k<256 :  n<512 -> Wq[k][n], else Wkv[k][n-512]
//   WtOut[n][k], n<256,  k<512 :  Wout[k][n]
// ---------------------------------------------------------------------------
__global__ void la_prep_kernel(const float* __restrict__ Wq,
                               const float* __restrict__ Wkv,
                               const float* __restrict__ Wout,
                               __bf16* __restrict__ WtQKV,
                               __bf16* __restrict__ WtOut) {
  int idx = blockIdx.x * 256 + threadIdx.x;
  if (idx < NQKV * DIM) {
    int n = idx >> 8, k = idx & 255;
    float v = (n < INNER) ? Wq[k * INNER + n] : Wkv[k * (2 * INNER) + (n - INNER)];
    WtQKV[n * DIM + k] = (__bf16)v;
  } else {
    int i2 = idx - NQKV * DIM;
    if (i2 < DIM * INNER) {
      int n = i2 >> 9, k = i2 & 511;
      WtOut[n * INNER + k] = (__bf16)Wout[k * DIM + n];
    }
  }
}

// ---------------------------------------------------------------------------
// Fused line attention: one workgroup (8 waves) per 24-row block.
// ---------------------------------------------------------------------------
__global__ __launch_bounds__(256)
void la_main_kernel(const float* __restrict__ x,
                    const __bf16* __restrict__ WtQKV,
                    const __bf16* __restrict__ WtOut,
                    const float* __restrict__ b_out,
                    const float* __restrict__ pos,
                    float* __restrict__ out) {
  extern __shared__ char smem[];
  __bf16* sX   = (__bf16*)(smem);
  __bf16* sQKV = (__bf16*)(smem + SX_BYTES);
  __bf16* sVt  = (__bf16*)(smem + SX_BYTES + SQKV_BYTES);
  float*  sStage = (float*)(smem + SX_BYTES + SQKV_BYTES);   // aliases sVt

  const int bId  = blockIdx.x;
  const int tid  = threadIdx.x;
  const int lane = tid & 31;
  const int w    = tid >> 5;     // wave id == head id
  const int ln   = lane & 15;
  const int lh   = lane >> 4;

  // warm L2 / L0 with this wave's weight slice
  __builtin_prefetch(WtQKV + (size_t)(w * 12 * 16) * DIM, 0, 3);

  // ---- Phase 0: x tile -> LDS -------------------------------------------
#if defined(__gfx1250__) && __has_builtin(__builtin_amdgcn_tensor_load_to_lds)
  // Tensor Data Mover: DMA the 24x256 fp32 tile into LDS staging (one wave
  // issues; tracked by TENSORcnt), then convert to bf16 into sX.
  if (w == 0) {
    unsigned long long ga = (unsigned long long)(const void*)(x + (size_t)bId * LINE * DIM);
    unsigned lds = (unsigned)(uintptr_t)(void*)sStage;
    u32x4 g0;
    g0[0] = 1u;                                      // count=1, user mode
    g0[1] = lds;                                     // lds_addr (bytes)
    g0[2] = (unsigned)(ga & 0xffffffffu);            // global_addr[31:0]
    g0[3] = (unsigned)((ga >> 32) & 0x01ffffffu)     // global_addr[56:32]
            | (2u << 30);                            // type=2 ("image")
    i32x8 g1;
    g1[0] = (int)(2u << 16);                         // data_size=2 -> 4 bytes
    g1[1] = (int)(DIM  << 16);                       // tensor_dim0[15:0]
    g1[2] = (int)(LINE << 16);                       // tensor_dim1[15:0]
    g1[3] = (int)(DIM  << 16);                       // tile_dim0
    g1[4] = LINE;                                    // tile_dim1 (tile_dim2=0)
    g1[5] = DIM;                                     // tensor_dim0_stride[31:0]
    g1[6] = 0;
    g1[7] = 0;
    i32x4 gz = {0, 0, 0, 0};
#if __clang_major__ >= 23
    i32x8 gz8 = {0, 0, 0, 0, 0, 0, 0, 0};
    __builtin_amdgcn_tensor_load_to_lds(g0, g1, gz, gz, gz8, 0);
#else
    __builtin_amdgcn_tensor_load_to_lds(g0, g1, gz, gz, 0);
#endif
    __builtin_amdgcn_s_wait_tensorcnt(0);
  }
  __syncthreads();
  for (int t = tid; t < 32 * (DIM / 4); t += 256) {
    int i  = t >> 6;
    int c4 = (t & 63) << 2;
    float4 v = make_float4(0.f, 0.f, 0.f, 0.f);
    if (i < LINE) v = *(const float4*)(sStage + i * DIM + c4);
    uint2 p;
    p.x = pk2(v.x, v.y);
    p.y = pk2(v.z, v.w);
    *(uint2*)(sX + i * PX + c4) = p;
  }
#else
  {
    const float* xb = x + (size_t)bId * LINE * DIM;
    for (int t = tid; t < 32 * (DIM / 4); t += 256) {
      int i  = t >> 6;
      int c4 = (t & 63) << 2;
      float4 v = make_float4(0.f, 0.f, 0.f, 0.f);
      if (i < LINE) v = *(const float4*)(xb + i * DIM + c4);
      uint2 p;
      p.x = pk2(v.x, v.y);
      p.y = pk2(v.z, v.w);
      *(uint2*)(sX + i * PX + c4) = p;
    }
  }
#endif
  __syncthreads();

  // ---- Phase 1: qkv^T GEMM.  D[m=col][n=i] tiles; wave w owns cols
  //      [192w, 192w+192).  q (cols<512) pre-scaled by dhead^-0.5. --------
  const float scale = 0.125f;
  for (int mt = 0; mt < 12; ++mt) {
    int n0 = (w * 12 + mt) << 4;
    f32x8 acc0 = zero8(), acc1 = zero8();
#pragma unroll
    for (int kk = 0; kk < 8; ++kk) {
      bf16x16 a  = frag_a(WtQKV, n0, DIM, kk * 32, lane);
      bf16x16 b0 = frag_b(sX, 0,  PX, kk * 32, lane);
      bf16x16 b1 = frag_b(sX, 16, PX, kk * 32, lane);
      acc0 = wmma_bf16(a, b0, acc0);
      acc1 = wmma_bf16(a, b1, acc1);
    }
    float s = (n0 < INNER) ? scale : 1.0f;
    int col0 = n0 + (lh << 3);
    *(uint4*)(sQKV + ln * PQ + col0)        = pack8(acc0, s);
    *(uint4*)(sQKV + (ln + 16) * PQ + col0) = pack8(acc1, s);
  }
  __syncthreads();

  // ---- Phase 2: transpose V region into sVt[hd][j] ----------------------
  for (int t = tid; t < INNER * 32; t += 256) {
    int hd = t >> 5, j = t & 31;
    sVt[hd * PV + j] = sQKV[j * PQ + 2 * INNER + hd];
  }
  __syncthreads();

  // ---- Phase 3: attention, head h = w ----------------------------------
  const int h = w;
  f32x8 sim[2][2];                              // [jt][it], simT[j][i]
  sim[0][0] = zero8(); sim[0][1] = zero8();
  sim[1][0] = zero8(); sim[1][1] = zero8();
  {
    bf16x16 af[2][2], qf[2][2];
#pragma unroll
    for (int jt = 0; jt < 2; ++jt)
#pragma unroll
      for (int kk = 0; kk < 2; ++kk)
        af[jt][kk] = frag_a(sQKV, jt * 16, PQ, INNER + h * DHEAD + kk * 32, lane);
#pragma unroll
    for (int it = 0; it < 2; ++it)
#pragma unroll
      for (int kk = 0; kk < 2; ++kk)
        qf[it][kk] = frag_b(sQKV, it * 16, PQ, h * DHEAD + kk * 32, lane);
#pragma unroll
    for (int it = 0; it < 2; ++it)
#pragma unroll
      for (int jt = 0; jt < 2; ++jt)
#pragma unroll
        for (int kk = 0; kk < 2; ++kk)
          sim[jt][it] = wmma_bf16(af[jt][kk], qf[it][kk], sim[jt][it]);
  }

  // pos_emb add + in-register softmax over j; build attn B-fragments.
  const float NEGBIG = -1e30f;
  unsigned Bfrag[2][8];
#pragma unroll
  for (int it = 0; it < 2; ++it) {
    int i    = ln + it * 16;
    int irow = (i < LINE) ? i : (LINE - 1);
    const float* prow = pos + (size_t)(h * LINE + irow) * LINE;
    float v0[8], v1[8];
    float4 pa = *(const float4*)(prow + lh * 8);
    float4 pb = *(const float4*)(prow + lh * 8 + 4);
    v0[0] = sim[0][it][0] + pa.x;  v0[1] = sim[0][it][1] + pa.y;
    v0[2] = sim[0][it][2] + pa.z;  v0[3] = sim[0][it][3] + pa.w;
    v0[4] = sim[0][it][4] + pb.x;  v0[5] = sim[0][it][5] + pb.y;
    v0[6] = sim[0][it][6] + pb.z;  v0[7] = sim[0][it][7] + pb.w;
    if (lh == 0) {                         // j = 16..23 valid
      float4 pc = *(const float4*)(prow + 16);
      float4 pd = *(const float4*)(prow + 20);
      v1[0] = sim[1][it][0] + pc.x;  v1[1] = sim[1][it][1] + pc.y;
      v1[2] = sim[1][it][2] + pc.z;  v1[3] = sim[1][it][3] + pc.w;
      v1[4] = sim[1][it][4] + pd.x;  v1[5] = sim[1][it][5] + pd.y;
      v1[6] = sim[1][it][6] + pd.z;  v1[7] = sim[1][it][7] + pd.w;
    } else {                               // j = 24..31 masked
#pragma unroll
      for (int r = 0; r < 8; ++r) v1[r] = NEGBIG;
    }
    float m = v0[0];
#pragma unroll
    for (int r = 1; r < 8; ++r) m = fmaxf(m, v0[r]);
#pragma unroll
    for (int r = 0; r < 8; ++r) m = fmaxf(m, v1[r]);
    m = fmaxf(m, __shfl_xor(m, 16, 32));
    float e0[8], e1[8], s = 0.f;
#pragma unroll
    for (int r = 0; r < 8; ++r) { e0[r] = __expf(v0[r] - m); s += e0[r]; }
#pragma unroll
    for (int r = 0; r < 8; ++r) {
      e1[r] = (lh == 0) ? __expf(v1[r] - m) : 0.f;
      s += e1[r];
    }
    s += __shfl_xor(s, 16, 32);
    float inv = 1.f / s;
    unsigned P0[4], P1[4];
#pragma unroll
    for (int p = 0; p < 4; ++p) {
      P0[p] = pk2(e0[2 * p] * inv, e0[2 * p + 1] * inv);
      P1[p] = pk2(e1[2 * p] * inv, e1[2 * p + 1] * inv);
    }
#pragma unroll
    for (int p = 0; p < 4; ++p) {
      unsigned o0 = __shfl_xor(P0[p], 16, 32);
      unsigned o1 = __shfl_xor(P1[p], 16, 32);
      Bfrag[it][p]     = lh ? o1    : P0[p];   // K chunk [16h, 16h+8)
      Bfrag[it][4 + p] = lh ? P1[p] : o0;      // K chunk [16h+8, 16h+16)
    }
  }

  // out^T[hd][i] = vT * attn^T; packed-store into dead q-region of sQKV.
#pragma unroll
  for (int mt = 0; mt < 4; ++mt) {
    bf16x16 av = frag_a(sVt, h * DHEAD + mt * 16, PV, 0, lane);
#pragma unroll
    for (int it = 0; it < 2; ++it) {
      FragU bu;
      bu.q[0] = make_uint4(Bfrag[it][0], Bfrag[it][1], Bfrag[it][2], Bfrag[it][3]);
      bu.q[1] = make_uint4(Bfrag[it][4], Bfrag[it][5], Bfrag[it][6], Bfrag[it][7]);
      f32x8 o = wmma_bf16(av, bu.v, zero8());
      int i    = ln + it * 16;
      int col0 = h * DHEAD + mt * 16 + (lh << 3);
      *(uint4*)(sQKV + i * PQ + col0) = pack8(o, 1.0f);
    }
  }
  __syncthreads();

  // ---- Phase 4: output projection.  finalT[c][i]; wave owns 2 c-tiles. --
  for (int mt2 = 0; mt2 < 2; ++mt2) {
    int c0t = (w * 2 + mt2) << 4;
    f32x8 acc0 = zero8(), acc1 = zero8();
#pragma unroll
    for (int kk = 0; kk < 16; ++kk) {
      bf16x16 a  = frag_a(WtOut, c0t, INNER, kk * 32, lane);
      bf16x16 b0 = frag_b(sQKV, 0,  PQ, kk * 32, lane);
      bf16x16 b1 = frag_b(sQKV, 16, PQ, kk * 32, lane);
      acc0 = wmma_bf16(a, b0, acc0);
      acc1 = wmma_bf16(a, b1, acc1);
    }
    int cbase = c0t + (lh << 3);
    float4 ba = *(const float4*)(b_out + cbase);
    float4 bb = *(const float4*)(b_out + cbase + 4);
    {   // it = 0, rows i = ln (< 16, always valid)
      float* orow = out + ((size_t)bId * LINE + ln) * DIM + cbase;
      float4 r0 = make_float4(acc0[0] + ba.x, acc0[1] + ba.y,
                              acc0[2] + ba.z, acc0[3] + ba.w);
      float4 r1 = make_float4(acc0[4] + bb.x, acc0[5] + bb.y,
                              acc0[6] + bb.z, acc0[7] + bb.w);
      *(float4*)(orow)     = r0;
      *(float4*)(orow + 4) = r1;
    }
    if (ln < 8) {  // it = 1, rows i = 16 + ln, valid for i < 24
      float* orow = out + ((size_t)bId * LINE + 16 + ln) * DIM + cbase;
      float4 r0 = make_float4(acc1[0] + ba.x, acc1[1] + ba.y,
                              acc1[2] + ba.z, acc1[3] + ba.w);
      float4 r1 = make_float4(acc1[4] + bb.x, acc1[5] + bb.y,
                              acc1[6] + bb.z, acc1[7] + bb.w);
      *(float4*)(orow)     = r0;
      *(float4*)(orow + 4) = r1;
    }
  }
}

extern "C" void kernel_launch(void* const* d_in, const int* in_sizes, int n_in,
                              void* d_out, int out_size, void* d_ws, size_t ws_size,
                              hipStream_t stream) {
  const float* x    = (const float*)d_in[0];
  const float* Wq   = (const float*)d_in[1];
  const float* Wkv  = (const float*)d_in[2];
  const float* Wout = (const float*)d_in[3];
  const float* bo   = (const float*)d_in[4];
  const float* pos  = (const float*)d_in[5];

  __bf16* WtQKV = (__bf16*)d_ws;                                   // 1536*256 bf16
  __bf16* WtOut = (__bf16*)((char*)d_ws + (size_t)NQKV * DIM * 2); // 256*512 bf16

  int prep_elems = NQKV * DIM + DIM * INNER;   // 524288
  la_prep_kernel<<<prep_elems / 256, 256, 0, stream>>>(Wq, Wkv, Wout, WtQKV, WtOut);

  la_main_kernel<<<NB, 256, LDS_BYTES, stream>>>(x, WtQKV, WtOut, bo, pos,
                                                 (float*)d_out);
}
